// GRUModel_79001628443426
// MI455X (gfx1250) — matrix operand order repaired and checked by reading, not compiled, not run on hardware
//
#include <hip/hip_runtime.h>
#include <hip/hip_bf16.h>
#include <math.h>

// Problem dims (fixed by setup_inputs)
#define TT 1024
#define BB 256
#define HH 256
#define CC 32
#define NWG 16          // HH / 16 column chunks, one WGP each

typedef __attribute__((ext_vector_type(16))) __bf16 v16bf;
typedef __attribute__((ext_vector_type(8)))  float  v8f;

union Frag32B { v16bf v; uint4 q[2]; };

__device__ __forceinline__ unsigned short f2bf(float f) {
    unsigned u = __float_as_uint(f);
    u += 0x7FFFu + ((u >> 16) & 1u);           // round-to-nearest-even
    return (unsigned short)(u >> 16);
}
__device__ __forceinline__ float fast_sigmoid(float x) {
    return 1.0f / (1.0f + __expf(-x));                 // v_exp + v_add + v_rcp
}
__device__ __forceinline__ float fast_tanh(float x) {
    return 1.0f - 2.0f / (1.0f + __expf(2.0f * x));    // v_exp based, no libm branch
}

// ---- device-scope sense barrier across the 16 persistent workgroups ----
__device__ __forceinline__ void grid_barrier(unsigned* cnt, unsigned* gen,
                                             unsigned nwg, unsigned& lgen) {
    __threadfence();
    __syncthreads();
    if (threadIdx.x == 0) {
        unsigned arrived = __hip_atomic_fetch_add(cnt, 1u, __ATOMIC_ACQ_REL,
                                                  __HIP_MEMORY_SCOPE_AGENT);
        if (arrived == nwg - 1u) {
            __hip_atomic_store(cnt, 0u, __ATOMIC_RELAXED, __HIP_MEMORY_SCOPE_AGENT);
            __hip_atomic_fetch_add(gen, 1u, __ATOMIC_RELEASE, __HIP_MEMORY_SCOPE_AGENT);
        } else {
            while (__hip_atomic_load(gen, __ATOMIC_ACQUIRE,
                                     __HIP_MEMORY_SCOPE_AGENT) == lgen) {
                __builtin_amdgcn_s_sleep(2);
            }
        }
        lgen++;
    }
    __syncthreads();
    __threadfence();
}

__global__ void init_bar(unsigned* bar) {
    if (threadIdx.x == 0) { bar[0] = 0u; bar[1] = 0u; }
}

__global__ __launch_bounds__(256)
void ajrnn_kernel(const float* __restrict__ x,      // [T,B,1]
                  const float* __restrict__ W_ih,   // [3H,1]
                  const float* __restrict__ b_ih,   // [3H]
                  const float* __restrict__ W_hh,   // [3H,H]
                  const float* __restrict__ b_hh,   // [3H]
                  const float* __restrict__ Wimp,   // [H,1]
                  const float* __restrict__ bias,   // [1]
                  const float* __restrict__ h0,     // [B,H]
                  const float* __restrict__ lin_w,  // [C,H]
                  const float* __restrict__ lin_b,  // [C]
                  float* __restrict__ out,          // hs | prediction | logits
                  unsigned short* __restrict__ hbuf0,
                  unsigned short* __restrict__ hbuf1,
                  unsigned* __restrict__ bar)
{
    __shared__ unsigned short sW[3 * 16 * HH];   // this WG's W_hh gate slice, bf16 (24 KB)
    __shared__ unsigned short sWimp[HH];         // imputation weight W, bf16 (512 B)

    const int tid  = threadIdx.x;
    const int wg   = blockIdx.x;                 // 0..15: owns h columns [wg*16, wg*16+16)
    const int c0   = wg * 16;
    const int wv   = tid >> 5;                   // wave 0..7 -> b-tiles {2w, 2w+1}
    const int lane = tid & 31;
    const int mb   = lane & 15;
    const int hi   = lane >> 4;

    // ---- one-time init: W_hh slice (3 gates x 16 rows x 256 k) -> bf16 LDS ----
    for (int i = tid; i < 3 * 16 * HH; i += 256) {
        int gate = i / (16 * HH);
        int rr   = (i / HH) & 15;
        int k    = i & (HH - 1);
        sW[i] = f2bf(W_hh[(size_t)(gate * HH + c0 + rr) * HH + k]);
    }
    sWimp[tid] = f2bf(Wimp[tid]);
    const float bias0 = bias[0];

    // h0 -> bf16 ping buffer (each WG converts its 16 rows)
    {
        int row  = wg * 16 + (tid >> 4);
        int colb = (tid & 15) * 16;
        for (int k = 0; k < 16; ++k)
            hbuf0[row * HH + colb + k] = f2bf(h0[row * HH + colb + k]);
    }

    // per-lane gate constants for column j = c0 + mb
    const int   j    = c0 + mb;
    const float bhr  = b_hh[j], bhz = b_hh[HH + j], bhn = b_hh[2 * HH + j];
    const float bir  = b_ih[j], biz = b_ih[HH + j], bin_ = b_ih[2 * HH + j];
    const float wir  = W_ih[j], wiz = W_ih[HH + j], win  = W_ih[2 * HH + j];

    const int arow0 = (2 * wv)     * 16 + mb;    // A-fragment row (b index) per lane
    const int arow1 = (2 * wv + 1) * 16 + mb;

    // h_old for (b, j) carried in registers: seeded from fp32 h0
    float hold[2][8];
    #pragma unroll
    for (int i = 0; i < 2; ++i)
        #pragma unroll
        for (int e = 0; e < 8; ++e)
            hold[i][e] = h0[(size_t)((2 * wv + i) * 16 + e + hi * 8) * HH + j];

    unsigned lgen = 0;
    grid_barrier(bar, bar + 1, gridDim.x, lgen);

    const size_t pred_off  = (size_t)TT * BB * HH;
    const size_t logit_off = pred_off + (size_t)BB * (TT - 1);

    for (int t = 0; t < TT; ++t) {
        const unsigned short* hprev = (t & 1) ? hbuf1 : hbuf0;
        unsigned short*       hnext = (t & 1) ? hbuf0 : hbuf1;

        // ---- preload this step's raw inputs for our 16 rows (overlaps the WMMAs) ----
        float xv[2][8];
        #pragma unroll
        for (int i = 0; i < 2; ++i)
            #pragma unroll
            for (int e = 0; e < 8; ++e)
                xv[i][e] = x[(size_t)t * BB + (2 * wv + i) * 16 + e + hi * 8];

        // ---- phase 2: gh (3 gates) AND x_hat via v_wmma_f32_16x16x32_bf16 ----
        // acc[i][g]  : gate pre-activations, C preloaded with b_hh
        // acc4[i]    : x_hat = h_prev @ W (+bias), B-fragment = W broadcast over columns
        v8f acc[2][3], acc4[2];
        #pragma unroll
        for (int e = 0; e < 8; ++e) {
            acc[0][0][e] = bhr; acc[0][1][e] = bhz; acc[0][2][e] = bhn;
            acc[1][0][e] = bhr; acc[1][1][e] = bhz; acc[1][2][e] = bhn;
            acc4[0][e] = bias0; acc4[1][e] = bias0;
        }
        #pragma unroll
        for (int k0 = 0; k0 < HH; k0 += 32) {
            Frag32B a0, a1, bw[3], bwp;
            // A 16x32 bf16, k order chosen here (B matches): lane<16 -> K k0..k0+15, lane>=16 -> K k0+16..k0+31
            const uint4* p0 = (const uint4*)(hprev + (size_t)arow0 * HH + k0 + hi * 16);
            const uint4* p1 = (const uint4*)(hprev + (size_t)arow1 * HH + k0 + hi * 16);
            a0.q[0] = p0[0]; a0.q[1] = p0[1];
            a1.q[0] = p1[0]; a1.q[1] = p1[1];
            // B 32x16: lane n holds column n = contiguous W_hh row, K = k0+hi*16 .. +15
            #pragma unroll
            for (int g = 0; g < 3; ++g) {
                const uint4* wp = (const uint4*)(sW + (g * 16 + mb) * HH + k0 + hi * 16);
                bw[g].q[0] = wp[0]; bw[g].q[1] = wp[1];
            }
            // Imputation B-fragment: every column n = W[k] (same 32B for all lanes in half)
            {
                const uint4* wp = (const uint4*)(sWimp + k0 + hi * 16);
                bwp.q[0] = wp[0]; bwp.q[1] = wp[1];
            }
            #pragma unroll
            for (int g = 0; g < 3; ++g) {
                acc[0][g] = __builtin_amdgcn_wmma_f32_16x16x32_bf16(
                    false, a0.v, false, bw[g].v, (short)0, acc[0][g], false, false);
                acc[1][g] = __builtin_amdgcn_wmma_f32_16x16x32_bf16(
                    false, a1.v, false, bw[g].v, (short)0, acc[1][g], false, false);
            }
            acc4[0] = __builtin_amdgcn_wmma_f32_16x16x32_bf16(
                false, a0.v, false, bwp.v, (short)0, acc4[0], false, false);
            acc4[1] = __builtin_amdgcn_wmma_f32_16x16x32_bf16(
                false, a1.v, false, bwp.v, (short)0, acc4[1], false, false);
        }

        // ---- phase 3: fused gates; D layout: b = bt*16 + e + 8*hi, col = c0 + mb ----
        float* hs_t = out + (size_t)t * BB * HH;
        const bool impute = (t > 0);
        #pragma unroll
        for (int i = 0; i < 2; ++i) {
            const int btb = (2 * wv + i) * 16;
            #pragma unroll
            for (int e = 0; e < 8; ++e) {
                const int   b    = btb + e + hi * 8;
                const float xraw = xv[i][e];
                const float cur  = (impute && xraw == 128.0f) ? acc4[i][e] : xraw;
                const float r    = fast_sigmoid(cur * wir + bir + acc[i][0][e]);
                const float z    = fast_sigmoid(cur * wiz + biz + acc[i][1][e]);
                const float n    = fast_tanh   (cur * win + bin_ + r * acc[i][2][e]);
                const float hnew = (1.0f - z) * n + z * hold[i][e];
                hold[i][e] = hnew;
                hs_t[(size_t)b * HH + j]  = hnew;
                hnext[(size_t)b * HH + j] = f2bf(hnew);
            }
        }
        grid_barrier(bar, bar + 1, gridDim.x, lgen);
    }

    // ---- epilogue: label_logits = hs[T-1] @ lin_w.T + lin_b (WG 0, fp32 from out) ----
    if (wg == 0) {
        const float* hf = out + (size_t)(TT - 1) * BB * HH;
        const int b = tid;
        for (int c = 0; c < CC; ++c) {
            float a = lin_b[c];
            const float4* hr = (const float4*)(hf + (size_t)b * HH);
            const float4* wr = (const float4*)(lin_w + (size_t)c * HH);
            #pragma unroll 8
            for (int kk = 0; kk < HH / 4; ++kk) {
                float4 hq = hr[kk], wq = wr[kk];
                a += hq.x * wq.x + hq.y * wq.y + hq.z * wq.z + hq.w * wq.w;
            }
            out[logit_off + (size_t)b * CC + c] = a;
        }
    }
}

// prediction[b, t] = hs[t, b] @ W + bias  for t in [0, T-2]  (fp32, fully parallel)
__global__ __launch_bounds__(256)
void pred_kernel(const float* __restrict__ hs,     // == out (hs region)
                 const float* __restrict__ Wimp,
                 const float* __restrict__ bias,
                 float* __restrict__ pred)
{
    __shared__ float sw[HH];
    sw[threadIdx.x] = Wimp[threadIdx.x];
    __syncthreads();
    const float b0 = bias[0];
    const int idx = blockIdx.x * 256 + threadIdx.x;   // idx = t*B + b
    if (idx < (TT - 1) * BB) {
        const int t = idx >> 8;          // BB == 256
        const int b = idx & (BB - 1);
        const float4* hr = (const float4*)(hs + (size_t)t * BB * HH + (size_t)b * HH);
        float a = b0;
        #pragma unroll 8
        for (int kk = 0; kk < HH / 4; ++kk) {
            float4 q = hr[kk];
            a += q.x * sw[4 * kk] + q.y * sw[4 * kk + 1] +
                 q.z * sw[4 * kk + 2] + q.w * sw[4 * kk + 3];
        }
        pred[(size_t)b * (TT - 1) + t] = a;
    }
}

extern "C" void kernel_launch(void* const* d_in, const int* in_sizes, int n_in,
                              void* d_out, int out_size, void* d_ws, size_t ws_size,
                              hipStream_t stream) {
    const float* x     = (const float*)d_in[0];
    const float* W_ih  = (const float*)d_in[1];
    const float* b_ih  = (const float*)d_in[2];
    const float* W_hh  = (const float*)d_in[3];
    const float* b_hh  = (const float*)d_in[4];
    const float* Wimp  = (const float*)d_in[5];
    const float* bias  = (const float*)d_in[6];
    const float* h0    = (const float*)d_in[7];
    const float* lin_w = (const float*)d_in[8];
    const float* lin_b = (const float*)d_in[9];

    char* ws = (char*)d_ws;
    unsigned short* hbuf0 = (unsigned short*)ws;                       // [B,H] bf16 ping
    unsigned short* hbuf1 = hbuf0 + (size_t)BB * HH;                   // [B,H] bf16 pong
    unsigned* bar = (unsigned*)(ws + 2 * (size_t)BB * HH * sizeof(unsigned short));

    float* out   = (float*)d_out;
    float* pred  = out + (size_t)TT * BB * HH;

    init_bar<<<1, 32, 0, stream>>>(bar);
    ajrnn_kernel<<<NWG, 256, 0, stream>>>(x, W_ih, b_ih, W_hh, b_hh, Wimp, bias,
                                          h0, lin_w, lin_b,
                                          out, hbuf0, hbuf1, bar);
    pred_kernel<<<((TT - 1) * BB + 255) / 256, 256, 0, stream>>>(out, Wimp, bias, pred);
}
